// ContextualSPROUTLayer_32865089749379
// MI455X (gfx1250) — matrix operand order repaired and checked by reading, not compiled, not run on hardware
//
#include <hip/hip_runtime.h>
#include <hip/hip_bf16.h>
#include <math.h>

typedef unsigned short u16;
typedef unsigned int   u32;
typedef __bf16         bf16_t;
typedef __attribute__((ext_vector_type(16))) bf16_t v16bf;
typedef __attribute__((ext_vector_type(8)))  float  v8f;

#define D_MODEL 1024
#define POOL    2048
#define TOPK    16
#define D_FF    4096
#define N_HEADS 16
#define HD      64
#define SEQ     1024
#define BATCH   4
#define NTOK    (BATCH*SEQ)

static __device__ __forceinline__ u16 f2bf_rne(float f) {
  u32 u = __builtin_bit_cast(u32, f);
  u32 r = (u + 0x7FFFu + ((u >> 16) & 1u)) >> 16;
  return (u16)r;
}
static __device__ __forceinline__ bf16_t bfbits(u16 v) {
  return __builtin_bit_cast(bf16_t, v);
}

struct frag_bits { uint4 lo; uint4 hi; };  // 32 bytes

// Build a v16bf fragment from two contiguous 16-byte runs.
static __device__ __forceinline__ v16bf ld_frag(const u16* plo, const u16* phi) {
  frag_bits fb;
  fb.lo = *(const uint4*)plo;
  fb.hi = *(const uint4*)phi;
  return __builtin_bit_cast(v16bf, fb);
}
// Fragment whose 16 elements are one contiguous 32-byte run.
static __device__ __forceinline__ v16bf ld_frag32(const u16* p) {
  return ld_frag(p, p + 8);
}

// ---------------------------------------------------------------------------
// fp32 -> bf16 (row-major), used for x only; weights are packed directly.
// ---------------------------------------------------------------------------
__global__ void k_f32_to_bf16(const float* __restrict__ in, u16* __restrict__ out, int n) {
  int i = blockIdx.x * blockDim.x + threadIdx.x;
  int stride = gridDim.x * blockDim.x;
  for (; i < n; i += stride) out[i] = f2bf_rne(in[i]);
}

// ---------------------------------------------------------------------------
// Pack fp32 weight B[K,N] into WMMA-B fragment-major bf16 layout:
// tile (kc, nt): 32 lanes x 16 elems contiguous; frag[2v]=B[kc*32+(v>>2)*16+h*8+2*(v&3)][n].
// One u32 (element pair) per thread iteration.
// ---------------------------------------------------------------------------
__global__ void k_pack_weight(const float* __restrict__ B, u16* __restrict__ out,
                              int N, int Kd) {
  const int total = (Kd * N) >> 1;
  const int ntiles = N >> 4;
  int i = blockIdx.x * blockDim.x + threadIdx.x;
  int stride = gridDim.x * blockDim.x;
  for (; i < total; i += stride) {
    int v = i & 7;
    int l = (i >> 3) & 31;
    int rest = i >> 8;
    int nt = rest % ntiles;
    int kc = rest / ntiles;
    int n = nt * 16 + (l & 15);
    int k = kc * 32 + (v >> 2) * 16 + (l >> 4) * 8 + 2 * (v & 3);
    u32 lo = f2bf_rne(B[(size_t)k * N + n]);
    u32 hi = f2bf_rne(B[(size_t)(k + 1) * N + n]);
    ((u32*)out)[i] = lo | (hi << 16);
  }
}

// ---------------------------------------------------------------------------
// WMMA bf16 GEMM: C[M,N] = A[M,K] @ B[K,N] + bias.
// A: row-major bf16 (fragments = two b128 loads). B: packed fragment-major.
// One wave -> 32x64 tile (2 A frags x 4 B frags = 8 wmma per k-step).
// Block = 4 waves -> 128x64. Grid = (N/64, M/128).
// ---------------------------------------------------------------------------
__global__ void k_gemm_bf16(const u16* __restrict__ A, const u16* __restrict__ Bp,
                            const float* __restrict__ bias,
                            float* __restrict__ C, u16* __restrict__ Cbf,
                            int M, int N, int Kd) {
  const int lane = threadIdx.x & 31;
  const int wave = threadIdx.x >> 5;
  const int m0 = blockIdx.y * 128 + wave * 32;
  const int n0 = blockIdx.x * 64;
  const int mm = lane & 15;
  const int hh = (lane >> 4) & 1;

  v8f acc[2][4] = {};
  const u16* Arow0 = A + (size_t)(m0 + mm) * Kd + hh * 8;
  const u16* Arow1 = Arow0 + (size_t)16 * Kd;
  const int ntiles = N >> 4;
  const u16* Bk = Bp + ((size_t)(n0 >> 4) * 32 + lane) * 16;
  const size_t bstep = (size_t)ntiles * 512;  // elements per k-chunk plane

  for (int k0 = 0; k0 < Kd; k0 += 32) {
    v16bf a0 = ld_frag(Arow0 + k0, Arow0 + k0 + 16);
    v16bf a1 = ld_frag(Arow1 + k0, Arow1 + k0 + 16);
    __builtin_prefetch((const void*)(Bk + bstep), 0, 3);
#pragma unroll
    for (int t = 0; t < 4; ++t) {
      v16bf bfr = ld_frag32(Bk + (size_t)t * 512);
      acc[0][t] = __builtin_amdgcn_wmma_f32_16x16x32_bf16(
          false, a0, false, bfr, (short)0, acc[0][t], false, false);
      acc[1][t] = __builtin_amdgcn_wmma_f32_16x16x32_bf16(
          false, a1, false, bfr, (short)0, acc[1][t], false, false);
    }
    Bk += bstep;
  }

#pragma unroll
  for (int t = 0; t < 4; ++t) {
    int nc = n0 + t * 16 + mm;
    float bv = bias ? bias[nc] : 0.f;
#pragma unroll
    for (int s = 0; s < 2; ++s) {
#pragma unroll
      for (int v = 0; v < 8; ++v) {
        int row = m0 + s * 16 + v + 8 * hh;
        float val = acc[s][t][v] + bv;
        if (C)   C[(size_t)row * N + nc] = val;
        if (Cbf) Cbf[(size_t)row * N + nc] = f2bf_rne(val);
      }
    }
  }
}

// ---------------------------------------------------------------------------
// Per-token top-16 (iterative argmax) + softmax. One block (256 thr) / token.
// ---------------------------------------------------------------------------
__global__ void k_topk_softmax(const float* __restrict__ scores,
                               float* __restrict__ wts, int* __restrict__ idx) {
  __shared__ float sc[POOL];
  __shared__ float rv[256];
  __shared__ int   ri[256];
  __shared__ float tv[TOPK];
  __shared__ int   ti[TOPK];
  const int t = blockIdx.x;
  const float* s = scores + (size_t)t * POOL;
  for (int i = threadIdx.x; i < POOL; i += 256) sc[i] = s[i];
  __syncthreads();

  for (int pass = 0; pass < TOPK; ++pass) {
    float best = -INFINITY; int bi = 0x7FFFFFFF;
    for (int i = threadIdx.x; i < POOL; i += 256) {
      float v = sc[i];
      if (v > best || (v == best && i < bi)) { best = v; bi = i; }
    }
    rv[threadIdx.x] = best; ri[threadIdx.x] = bi;
    __syncthreads();
    for (int off = 128; off > 0; off >>= 1) {
      if (threadIdx.x < off) {
        float ov = rv[threadIdx.x + off]; int oi = ri[threadIdx.x + off];
        if (ov > rv[threadIdx.x] || (ov == rv[threadIdx.x] && oi < ri[threadIdx.x])) {
          rv[threadIdx.x] = ov; ri[threadIdx.x] = oi;
        }
      }
      __syncthreads();
    }
    if (threadIdx.x == 0) { tv[pass] = rv[0]; ti[pass] = ri[0]; sc[ri[0]] = -INFINITY; }
    __syncthreads();
  }

  if (threadIdx.x == 0) {
    float mx = tv[0];
    float e[TOPK]; float sum = 0.f;
    for (int k = 0; k < TOPK; ++k) { e[k] = __expf(tv[k] - mx); sum += e[k]; }
    float inv = 1.f / sum;
    for (int k = 0; k < TOPK; ++k) {
      wts[(size_t)t * TOPK + k] = e[k] * inv;
      idx[(size_t)t * TOPK + k] = ti[k];
    }
  }
}

// ---------------------------------------------------------------------------
// Gather 16 firing rows, weighted combine, exact GELU, emit bf16.
// float4 loads / uint2 stores. One block (256 thr) / token.
// ---------------------------------------------------------------------------
__global__ void k_combine_gelu(const float* __restrict__ firing,
                               const float* __restrict__ wts, const int* __restrict__ idx,
                               u16* __restrict__ outbf) {
  __shared__ float w[TOPK];
  __shared__ int   id[TOPK];
  const int t = blockIdx.x;
  if (threadIdx.x < TOPK) {
    w[threadIdx.x]  = wts[(size_t)t * TOPK + threadIdx.x];
    id[threadIdx.x] = idx[(size_t)t * TOPK + threadIdx.x];
  }
  __syncthreads();

  float4 acc[D_FF / 1024];  // 4 float4 per thread
#pragma unroll
  for (int i = 0; i < D_FF / 1024; ++i) acc[i] = make_float4(0.f, 0.f, 0.f, 0.f);

  for (int k = 0; k < TOPK; ++k) {
    const float4* row = (const float4*)(firing + (size_t)id[k] * D_FF);
    float wk = w[k];
#pragma unroll
    for (int i = 0; i < D_FF / 1024; ++i) {
      float4 f = row[threadIdx.x + i * 256];
      acc[i].x += wk * f.x; acc[i].y += wk * f.y;
      acc[i].z += wk * f.z; acc[i].w += wk * f.w;
    }
  }
#pragma unroll
  for (int i = 0; i < D_FF / 1024; ++i) {
    float g0 = 0.5f * acc[i].x * (1.f + erff(acc[i].x * 0.70710678118654752f));
    float g1 = 0.5f * acc[i].y * (1.f + erff(acc[i].y * 0.70710678118654752f));
    float g2 = 0.5f * acc[i].z * (1.f + erff(acc[i].z * 0.70710678118654752f));
    float g3 = 0.5f * acc[i].w * (1.f + erff(acc[i].w * 0.70710678118654752f));
    u32 lo = (u32)f2bf_rne(g0) | ((u32)f2bf_rne(g1) << 16);
    u32 hi = (u32)f2bf_rne(g2) | ((u32)f2bf_rne(g3) << 16);
    u32* dst = (u32*)(outbf + (size_t)t * D_FF) + (threadIdx.x + i * 256) * 2;
    dst[0] = lo; dst[1] = hi;
  }
}

// ---------------------------------------------------------------------------
// Repack V from qkv_bf into WMMA-B fragment-major layout per
// (batch, head, key-chunk of 32, d-tile of 16): 32 lanes x 16 elems contiguous.
// ---------------------------------------------------------------------------
__global__ void k_pack_v(const u16* __restrict__ qkv, u16* __restrict__ out) {
  const int total = (NTOK * D_MODEL) >> 1;  // u32 count
  int i = blockIdx.x * blockDim.x + threadIdx.x;
  int stride = gridDim.x * blockDim.x;
  for (; i < total; i += stride) {
    int v    = i & 7;
    int l    = (i >> 3) & 31;
    int dt   = (i >> 8) & 3;
    int kc   = (i >> 10) & 31;
    int head = (i >> 15) & 15;
    int b    = i >> 19;
    int d    = dt * 16 + (l & 15);
    int key  = kc * 32 + (v >> 2) * 16 + (l >> 4) * 8 + 2 * (v & 3);
    const u16* src = qkv + (size_t)(b * SEQ + key) * (3 * D_MODEL)
                         + 2 * D_MODEL + head * HD + d;
    u32 lo = src[0];
    u32 hi = src[3 * D_MODEL];
    ((u32*)out)[i] = lo | (hi << 16);
  }
}

// ---------------------------------------------------------------------------
// Flash attention. One wave per 16-row q-tile. Q/K fragments are contiguous
// b128 loads from qkv_bf; V fragments from packed V; P transposed via LDS.
// ---------------------------------------------------------------------------
__global__ void k_attention(const u16* __restrict__ qkv, const u16* __restrict__ vpack,
                            u16* __restrict__ ctx) {
  __shared__ __align__(16) float plds[4][16 * 32];
  const int lane = threadIdx.x & 31;
  const int wv   = threadIdx.x >> 5;
  const int wid  = blockIdx.x * 4 + wv;
  const int qt   = wid & 63;
  const int head = (wid >> 6) & 15;
  const int b    = wid >> 10;
  const int mm   = lane & 15;
  const int hh   = (lane >> 4) & 1;
  float* P = plds[wv];

  // Q fragments (A-layout): two b128 loads per 32-wide d-chunk
  const u16* Qrow = qkv + (size_t)(b * SEQ + qt * 16 + mm) * (3 * D_MODEL)
                        + head * HD + hh * 8;
  v16bf aq[2];
#pragma unroll
  for (int c = 0; c < 2; ++c)
    aq[c] = ld_frag(Qrow + c * 32, Qrow + c * 32 + 16);

  v8f o[4] = {};
  float Mrow[8], Lrow[8];
#pragma unroll
  for (int v = 0; v < 8; ++v) { Mrow[v] = -INFINITY; Lrow[v] = 0.f; }

  const u16* Kbase = qkv + (size_t)b * SEQ * (3 * D_MODEL) + D_MODEL + head * HD;
  const u16* Vhead = vpack + (size_t)(b * N_HEADS + head) * (SEQ / 32) * 4 * 512
                           + (size_t)lane * 16;

  for (int j = 0; j < SEQ / 32; ++j) {
    // ---- scores: two 16x16 tiles (32 keys) ----
    v8f sacc[2] = {};
#pragma unroll
    for (int st = 0; st < 2; ++st) {
      int key = j * 32 + st * 16 + mm;
      const u16* Krow = Kbase + (size_t)key * (3 * D_MODEL) + hh * 8;
#pragma unroll
      for (int c = 0; c < 2; ++c) {
        v16bf bk = ld_frag(Krow + c * 32, Krow + c * 32 + 16);  // two b128 d-runs
        sacc[st] = __builtin_amdgcn_wmma_f32_16x16x32_bf16(
            false, aq[c], false, bk, (short)0, sacc[st], false, false);
      }
    }
    // ---- online softmax ----
    float nm[8];
#pragma unroll
    for (int v = 0; v < 8; ++v) {
      float s0 = sacc[0][v] * 0.125f;
      float s1 = sacc[1][v] * 0.125f;
      sacc[0][v] = s0; sacc[1][v] = s1;
      float mx = fmaxf(s0, s1);
#pragma unroll
      for (int msk = 1; msk < 16; msk <<= 1) mx = fmaxf(mx, __shfl_xor(mx, msk, 32));
      nm[v] = fmaxf(Mrow[v], mx);
    }
#pragma unroll
    for (int v = 0; v < 8; ++v) {
      float alpha = __expf(Mrow[v] - nm[v]);
      float p0 = __expf(sacc[0][v] - nm[v]);
      float p1 = __expf(sacc[1][v] - nm[v]);
      float rs = p0 + p1;
#pragma unroll
      for (int msk = 1; msk < 16; msk <<= 1) rs += __shfl_xor(rs, msk, 32);
      Lrow[v] = Lrow[v] * alpha + rs;
      Mrow[v] = nm[v];
#pragma unroll
      for (int t = 0; t < 4; ++t) o[t][v] *= alpha;
      int row = v + 8 * hh;
      P[row * 32 + mm]      = p0;
      P[row * 32 + 16 + mm] = p1;
    }
    asm volatile("s_wait_dscnt 0" ::: "memory");
    // ---- P (16x32) -> A-layout bf16 via float4 LDS reads ----
    float4 q0 = *(const float4*)&P[mm * 32 + hh * 8];
    float4 q1 = *(const float4*)&P[mm * 32 + hh * 8 + 4];
    float4 q2 = *(const float4*)&P[mm * 32 + 16 + hh * 8];
    float4 q3 = *(const float4*)&P[mm * 32 + 16 + hh * 8 + 4];
    v16bf pa;
    pa[0]  = bfbits(f2bf_rne(q0.x)); pa[1]  = bfbits(f2bf_rne(q0.y));
    pa[2]  = bfbits(f2bf_rne(q0.z)); pa[3]  = bfbits(f2bf_rne(q0.w));
    pa[4]  = bfbits(f2bf_rne(q1.x)); pa[5]  = bfbits(f2bf_rne(q1.y));
    pa[6]  = bfbits(f2bf_rne(q1.z)); pa[7]  = bfbits(f2bf_rne(q1.w));
    pa[8]  = bfbits(f2bf_rne(q2.x)); pa[9]  = bfbits(f2bf_rne(q2.y));
    pa[10] = bfbits(f2bf_rne(q2.z)); pa[11] = bfbits(f2bf_rne(q2.w));
    pa[12] = bfbits(f2bf_rne(q3.x)); pa[13] = bfbits(f2bf_rne(q3.y));
    pa[14] = bfbits(f2bf_rne(q3.z)); pa[15] = bfbits(f2bf_rne(q3.w));
    // ---- O += P @ V ----
    const u16* Vj = Vhead + (size_t)j * 4 * 512;
#pragma unroll
    for (int t = 0; t < 4; ++t) {
      v16bf bv = ld_frag32(Vj + (size_t)t * 512);
      o[t] = __builtin_amdgcn_wmma_f32_16x16x32_bf16(
          false, pa, false, bv, (short)0, o[t], false, false);
    }
  }

#pragma unroll
  for (int t = 0; t < 4; ++t) {
    int d = t * 16 + mm;
#pragma unroll
    for (int v = 0; v < 8; ++v) {
      int row = qt * 16 + v + 8 * hh;
      float val = o[t][v] / Lrow[v];
      ctx[(size_t)(b * SEQ + row) * D_MODEL + head * HD + d] = f2bf_rne(val);
    }
  }
}

// ---------------------------------------------------------------------------
// out = LN2( LN1(x + attn) + neuron ). One block (256 thr) / token.
// ---------------------------------------------------------------------------
__global__ void k_residual_ln(const float* __restrict__ x, const float* __restrict__ attn,
                              const float* __restrict__ neuron,
                              const float* __restrict__ g1, const float* __restrict__ b1,
                              const float* __restrict__ g2, const float* __restrict__ b2,
                              float* __restrict__ out) {
  __shared__ float red[256];
  __shared__ float red2[256];
  const int t = blockIdx.x;
  const float* xp = x + (size_t)t * D_MODEL;
  const float* ap = attn + (size_t)t * D_MODEL;
  const float* np = neuron + (size_t)t * D_MODEL;

  float v1[4];
  float s = 0.f, ss = 0.f;
#pragma unroll
  for (int i = 0; i < 4; ++i) {
    int c = threadIdx.x + i * 256;
    float val = xp[c] + ap[c];
    v1[i] = val; s += val; ss += val * val;
  }
  red[threadIdx.x] = s; red2[threadIdx.x] = ss;
  __syncthreads();
  for (int off = 128; off > 0; off >>= 1) {
    if (threadIdx.x < off) { red[threadIdx.x] += red[threadIdx.x + off]; red2[threadIdx.x] += red2[threadIdx.x + off]; }
    __syncthreads();
  }
  float mu = red[0] / D_MODEL;
  float var = red2[0] / D_MODEL - mu * mu;
  float rs = rsqrtf(var + 1e-5f);
  __syncthreads();

  float v2[4];
  s = 0.f; ss = 0.f;
#pragma unroll
  for (int i = 0; i < 4; ++i) {
    int c = threadIdx.x + i * 256;
    float h = (v1[i] - mu) * rs * g1[c] + b1[c];
    float val = h + np[c];
    v2[i] = val; s += val; ss += val * val;
  }
  red[threadIdx.x] = s; red2[threadIdx.x] = ss;
  __syncthreads();
  for (int off = 128; off > 0; off >>= 1) {
    if (threadIdx.x < off) { red[threadIdx.x] += red[threadIdx.x + off]; red2[threadIdx.x] += red2[threadIdx.x + off]; }
    __syncthreads();
  }
  float mu2 = red[0] / D_MODEL;
  float var2 = red2[0] / D_MODEL - mu2 * mu2;
  float rs2 = rsqrtf(var2 + 1e-5f);
#pragma unroll
  for (int i = 0; i < 4; ++i) {
    int c = threadIdx.x + i * 256;
    out[(size_t)t * D_MODEL + c] = (v2[i] - mu2) * rs2 * g2[c] + b2[c];
  }
}

// ---------------------------------------------------------------------------
extern "C" void kernel_launch(void* const* d_in, const int* in_sizes, int n_in,
                              void* d_out, int out_size, void* d_ws, size_t ws_size,
                              hipStream_t stream) {
  (void)in_sizes; (void)n_in; (void)out_size; (void)ws_size;
  const float* x          = (const float*)d_in[0];
  const float* w_router   = (const float*)d_in[1];
  const float* b_router   = (const float*)d_in[2];
  const float* firing     = (const float*)d_in[3];
  const float* w_pool_out = (const float*)d_in[4];
  const float* b_pool_out = (const float*)d_in[5];
  const float* w_in       = (const float*)d_in[6];
  const float* b_in       = (const float*)d_in[7];
  const float* w_attn_out = (const float*)d_in[8];
  const float* b_attn_out = (const float*)d_in[9];
  const float* g1         = (const float*)d_in[10];
  const float* beta1      = (const float*)d_in[11];
  const float* g2         = (const float*)d_in[12];
  const float* beta2      = (const float*)d_in[13];

  char* ws = (char*)d_ws;
  size_t off = 0;
  auto alloc = [&](size_t bytes) -> void* {
    void* p = ws + off;
    off += (bytes + 255) & ~(size_t)255;
    return p;
  };

  u16*   x_bf      = (u16*)alloc((size_t)NTOK * D_MODEL * 2);
  u16*   wr_pk     = (u16*)alloc((size_t)D_MODEL * POOL * 2);
  u16*   wp_pk     = (u16*)alloc((size_t)D_FF * D_MODEL * 2);
  u16*   wi_pk     = (u16*)alloc((size_t)D_MODEL * 3 * D_MODEL * 2);
  u16*   wa_pk     = (u16*)alloc((size_t)D_MODEL * D_MODEL * 2);
  float* scores    = (float*)alloc((size_t)NTOK * POOL * 4);
  float* topw      = (float*)alloc((size_t)NTOK * TOPK * 4);
  int*   topi      = (int*)alloc((size_t)NTOK * TOPK * 4);
  float* neuron_f  = (float*)alloc((size_t)NTOK * D_MODEL * 4);
  u16*   neuron_bf = (u16*)alloc((size_t)NTOK * D_MODEL * 2);
  u16*   qkv_bf    = (u16*)alloc((size_t)NTOK * 3 * D_MODEL * 2);
  u16*   v_pk      = (u16*)alloc((size_t)NTOK * D_MODEL * 2);
  u16*   ctx_bf    = (u16*)alloc((size_t)NTOK * D_MODEL * 2);
  float* attn_f    = (float*)alloc((size_t)NTOK * D_MODEL * 4);
  // combined (bf16, NTOK*D_FF) reuses the scores region (same byte size)
  u16*   comb_bf   = (u16*)scores;

  // 1) operand prep: x -> bf16; weights -> packed WMMA-B bf16
  k_f32_to_bf16<<<2048, 256, 0, stream>>>(x, x_bf, NTOK * D_MODEL);
  k_pack_weight<<<2048, 256, 0, stream>>>(w_router, wr_pk, POOL, D_MODEL);
  k_pack_weight<<<2048, 256, 0, stream>>>(w_pool_out, wp_pk, D_MODEL, D_FF);
  k_pack_weight<<<2048, 256, 0, stream>>>(w_in, wi_pk, 3 * D_MODEL, D_MODEL);
  k_pack_weight<<<2048, 256, 0, stream>>>(w_attn_out, wa_pk, D_MODEL, D_MODEL);

  // 2) router scores = x @ w_router + b_router
  k_gemm_bf16<<<dim3(POOL / 64, NTOK / 128), 128, 0, stream>>>(
      x_bf, wr_pk, b_router, scores, nullptr, NTOK, POOL, D_MODEL);

  // 3) top-16 + softmax
  k_topk_softmax<<<NTOK, 256, 0, stream>>>(scores, topw, topi);

  // 4) gather + weighted combine + GELU (bf16) — overwrites the scores region
  k_combine_gelu<<<NTOK, 256, 0, stream>>>(firing, topw, topi, comb_bf);

  // 5) neuron_out = gelu_combined @ w_pool_out + b
  k_gemm_bf16<<<dim3(D_MODEL / 64, NTOK / 128), 128, 0, stream>>>(
      comb_bf, wp_pk, b_pool_out, neuron_f, neuron_bf, NTOK, D_MODEL, D_FF);

  // 6) qkv = neuron_out @ w_in + b_in (bf16 only)
  k_gemm_bf16<<<dim3(3 * D_MODEL / 64, NTOK / 128), 128, 0, stream>>>(
      neuron_bf, wi_pk, b_in, nullptr, qkv_bf, NTOK, 3 * D_MODEL, D_MODEL);

  // 7) pack V fragment-major, then attention -> ctx (bf16)
  k_pack_v<<<2048, 256, 0, stream>>>(qkv_bf, v_pk);
  k_attention<<<(BATCH * N_HEADS * (SEQ / 16)) / 4, 128, 0, stream>>>(
      qkv_bf, v_pk, ctx_bf);

  // 8) attn_out = ctx @ w_attn_out + b
  k_gemm_bf16<<<dim3(D_MODEL / 64, NTOK / 128), 128, 0, stream>>>(
      ctx_bf, wa_pk, b_attn_out, attn_f, nullptr, NTOK, D_MODEL, D_MODEL);

  // 9) residual + LN1 + residual + LN2
  k_residual_ln<<<NTOK, 256, 0, stream>>>(x, attn_f, neuron_f,
                                          g1, beta1, g2, beta2, (float*)d_out);
}
